// GCN_28132035789324
// MI455X (gfx1250) — matrix operand order repaired and checked by reading, not compiled, not run on hardware
//
#include <hip/hip_runtime.h>
#include <hip/hip_bf16.h>

#define N_NODES 100000
#define N_EDGES 1600000
#define E_TOT   (N_EDGES + N_NODES)   // edges + self loops = 1,700,000
#define D_HID   64
#define D_IN    264
#define KP1     288                   // D_IN padded to multiple of 32
#define NEG_SLOPE 0.2f

typedef __attribute__((ext_vector_type(16))) __bf16 v16bf;
typedef __attribute__((ext_vector_type(8)))  __bf16 bf16x8;
typedef __attribute__((ext_vector_type(4)))  __bf16 bf16x4;
typedef __attribute__((ext_vector_type(8)))  float  v8f;

// ---- order-preserving float <-> uint key (for atomic max on floats) ----
__device__ __forceinline__ unsigned ordKey(float f) {
    unsigned u = __float_as_uint(f);
    return (u & 0x80000000u) ? ~u : (u | 0x80000000u);
}
__device__ __forceinline__ float ordVal(unsigned k) {
    unsigned u = (k & 0x80000000u) ? (k & 0x7FFFFFFFu) : ~k;
    return __uint_as_float(u);
}
#define ORD_NEG 0x00800000u   // == ordKey(-FLT_MAX)

// =====================================================================
// Init per-launch scratch (graph-replay safe: runs at head of every launch)
// =====================================================================
__global__ __launch_bounds__(256) void init_ws(unsigned* __restrict__ outu,
                                               unsigned* __restrict__ mxu,
                                               float* __restrict__ den,
                                               float* __restrict__ meanAcc) {
    int i = blockIdx.x * 256 + threadIdx.x;
    if (i < N_NODES * 64) outu[i] = ORD_NEG;
    if (i < N_NODES) { mxu[i] = ORD_NEG; den[i] = 0.0f; }
    if (i == 0) meanAcc[0] = 0.0f;
}

// mean(edge_attr): block-reduce + one atomic per block
__global__ __launch_bounds__(256) void mean_reduce(const float* __restrict__ ea,
                                                   float* __restrict__ acc) {
    __shared__ float sm[256];
    float s = 0.0f;
    for (int i = blockIdx.x * 256 + threadIdx.x; i < N_EDGES; i += gridDim.x * 256)
        s += ea[i];
    sm[threadIdx.x] = s;
    __syncthreads();
    for (int o = 128; o > 0; o >>= 1) {
        if (threadIdx.x < o) sm[threadIdx.x] += sm[threadIdx.x + o];
        __syncthreads();
    }
    if (threadIdx.x == 0) atomicAdd(acc, sm[0]);
}

// =====================================================================
// Pack a K x 64 fp32 weight into bf16, column-major, K padded to Kp.
// Output layout matches the WMMA B-fragment: lane (col) reads 16
// halfs at col*Kp + kt*32 + hi*8 + {0..7, 16..23}.
// =====================================================================
__global__ __launch_bounds__(256) void pack_weight(const float* __restrict__ W,
                                                   int K, int Kp,
                                                   __bf16* __restrict__ out) {
    int i = blockIdx.x * 256 + threadIdx.x;   // over 64*Kp
    if (i >= 64 * Kp) return;
    int col = i / Kp;
    int k   = i - col * Kp;
    out[i] = (k < K) ? (__bf16)W[(size_t)k * 64 + col] : (__bf16)0.0f;
}

// =====================================================================
// Dual-output GEMM: O0 = A@W0 + b0, O1 = A@W1 + b1  (A: N x K fp32,
// packed bf16 weights).  One block = 16 rows of A; 8 waves =
// {2 matrices} x {4 column tiles of 16}.  A tile async-loaded to LDS
// as fp32 (converted to bf16 while building fragments, co-executing
// with WMMA); B fragments are two global_load_b128 from packed weights.
// No barriers inside the k-loop.
// =====================================================================
__global__ __launch_bounds__(256) void gemm_dual(
    const float* __restrict__ A, int K, int Kp,
    const __bf16* __restrict__ Wb0, const float* __restrict__ b0, float* __restrict__ O0,
    const __bf16* __restrict__ Wb1, const float* __restrict__ b1, float* __restrict__ O1,
    int nmat, int doRelu)
{
    // Only shared object in this kernel -> starts at LDS offset 0,
    // so raw byte offsets can be handed to the async-load instruction.
    __shared__ __attribute__((aligned(16))) float Asf[16 * KP1];   // 18432 B

    const int tid     = threadIdx.x;
    const int rowBase = blockIdx.x * 16;
    const int kSteps  = Kp >> 5;
    const int sr      = tid >> 4;     // staging row   0..15
    const int sc      = tid & 15;     // staging chunk 0..15

    // zero-fill padded tail columns (K..Kp), if any
    for (int c = K + sc; c < Kp; c += 16) Asf[sr * KP1 + c] = 0.0f;

    // async-stage the 16 x K fp32 tile into LDS, 16 B per op, coalesced
    for (int c4 = sc * 4; c4 < K; c4 += 64) {
        unsigned lds = (unsigned)((sr * KP1 + c4) * 4);
        const float* gp = &A[(size_t)(rowBase + sr) * K + c4];
        asm volatile("global_load_async_to_lds_b128 %0, %1, off"
                     :: "v"(lds), "v"(gp) : "memory");
    }
    asm volatile("s_wait_asynccnt 0x0" ::: "memory");
    __syncthreads();

    const int lane = tid & 31;
    const int wv   = tid >> 5;
    const int mat  = wv >> 2;           // 0: W0, 1: W1
    const int ct   = wv & 3;            // column tile (16 cols each)
    const int rn   = lane & 15;
    const int hi   = lane >> 4;

    const __bf16* Wb   = (mat == 0 || nmat == 1) ? Wb0 : Wb1;
    const __bf16* wcol = Wb + (size_t)(ct * 16 + rn) * Kp;

    v8f acc = {0.f, 0.f, 0.f, 0.f, 0.f, 0.f, 0.f, 0.f};

    for (int kt = 0; kt < kSteps; ++kt) {
        // A fragment: row rn, K = kt*32 + hi*8 + {0..7, 16..23}
        const int k0 = kt * 32 + hi * 8;
        const float* ap = &Asf[rn * KP1 + k0];
        float4 f0 = *(const float4*)(ap);
        float4 f1 = *(const float4*)(ap + 4);
        float4 f2 = *(const float4*)(ap + 16);
        float4 f3 = *(const float4*)(ap + 20);
        v16bf a;
        a[0]  = (__bf16)f0.x; a[1]  = (__bf16)f0.y; a[2]  = (__bf16)f0.z; a[3]  = (__bf16)f0.w;
        a[4]  = (__bf16)f1.x; a[5]  = (__bf16)f1.y; a[6]  = (__bf16)f1.z; a[7]  = (__bf16)f1.w;
        a[8]  = (__bf16)f2.x; a[9]  = (__bf16)f2.y; a[10] = (__bf16)f2.z; a[11] = (__bf16)f2.w;
        a[12] = (__bf16)f3.x; a[13] = (__bf16)f3.y; a[14] = (__bf16)f3.z; a[15] = (__bf16)f3.w;

        // B fragment: K = kt*32 + hi*8 + {0..7, 16..23} from packed weight (two b128 loads)
        const __bf16* bp = wcol + kt * 32 + hi * 8;
        bf16x8 q0 = *(const bf16x8*)bp;
        bf16x8 q1 = *(const bf16x8*)(bp + 16);
        v16bf b = __builtin_shufflevector(q0, q1, 0,1,2,3,4,5,6,7,8,9,10,11,12,13,14,15);

        acc = __builtin_amdgcn_wmma_f32_16x16x32_bf16(false, a, false, b,
                                                      (short)0, acc, false, false);
    }

    // epilogue: C VGPR i holds M = i + 8*hi, N = lane%16
    const int col  = ct * 16 + rn;
    const float bias = ((mat == 0) ? b0 : b1)[col];
    float* O = (mat == 0) ? O0 : O1;
    if (mat == 0 || nmat == 2) {
        for (int i = 0; i < 8; ++i) {
            int row = rowBase + i + (hi << 3);
            float v = acc[i] + bias;
            if (doRelu) v = fmaxf(v, 0.0f);
            O[(size_t)row * 64 + col] = v;
        }
    }
}

// =====================================================================
// Edge pass 1: wave per edge. logits[e] = sum_k att[k] *
//   leaky_relu(xr[dst]+xl[src]+ea*We, 0.2); segment-max into mxu[dst].
// =====================================================================
__global__ __launch_bounds__(256) void edge_logits_max(
    const int* __restrict__ src, const int* __restrict__ dst,
    const float* __restrict__ ea, const float* __restrict__ meanAcc,
    const float* __restrict__ xl, const float* __restrict__ xr,
    const float* __restrict__ We, const float* __restrict__ att,
    float* __restrict__ logits, unsigned* __restrict__ mxu)
{
    int g = blockIdx.x * 256 + threadIdx.x;
    int e = g >> 5, lane = g & 31;
    if (e >= E_TOT) return;
    int s, d; float eav;
    if (e < N_EDGES) { s = src[e]; d = dst[e]; eav = ea[e]; }
    else             { s = d = e - N_EDGES; eav = meanAcc[0] * (1.0f / (float)N_EDGES); }

    const float2 vl = *(const float2*)(xl + (size_t)s * 64 + lane * 2);
    const float2 vr = *(const float2*)(xr + (size_t)d * 64 + lane * 2);
    const float2 we = *(const float2*)(We + lane * 2);
    const float2 at = *(const float2*)(att + lane * 2);

    float m0 = vr.x + vl.x + eav * we.x;
    float m1 = vr.y + vl.y + eav * we.y;
    m0 = (m0 > 0.0f) ? m0 : NEG_SLOPE * m0;
    m1 = (m1 > 0.0f) ? m1 : NEG_SLOPE * m1;
    float t = m0 * at.x + m1 * at.y;
    for (int o = 16; o > 0; o >>= 1) t += __shfl_xor(t, o);
    if (lane == 0) {
        logits[e] = t;
        atomicMax(&mxu[d], ordKey(t));
    }
}

// Edge pass 2: ex = exp(logit - mx[dst]); den[dst] += ex (in place)
__global__ __launch_bounds__(256) void edge_exp_sum(
    const int* __restrict__ dst, float* __restrict__ logits,
    const unsigned* __restrict__ mxu, float* __restrict__ den)
{
    int e = blockIdx.x * 256 + threadIdx.x;
    if (e >= E_TOT) return;
    int d = (e < N_EDGES) ? dst[e] : (e - N_EDGES);
    float ex = expf(logits[e] - ordVal(mxu[d]));
    logits[e] = ex;
    atomicAdd(&den[d], ex);
}

// Edge pass 3: alpha = ex/den[dst]; segment-MAX of xl[src]*alpha per feature
__global__ __launch_bounds__(256) void edge_aggr_max(
    const int* __restrict__ src, const int* __restrict__ dst,
    const float* __restrict__ exv, const float* __restrict__ den,
    const float* __restrict__ xl, unsigned* __restrict__ outu)
{
    int g = blockIdx.x * 256 + threadIdx.x;
    int e = g >> 5, lane = g & 31;
    if (e >= E_TOT) return;
    int s, d;
    if (e < N_EDGES) { s = src[e]; d = dst[e]; }
    else             { s = d = e - N_EDGES; }
    float alpha = exv[e] / den[d];
    float2 vl = *(const float2*)(xl + (size_t)s * 64 + lane * 2);
    unsigned* o = outu + (size_t)d * 64 + lane * 2;
    atomicMax(o,     ordKey(vl.x * alpha));
    atomicMax(o + 1, ordKey(vl.y * alpha));
}

// Node finalize: h = relu(segmax + bias); reset scratch for next layer
__global__ __launch_bounds__(256) void node_finalize(
    unsigned* __restrict__ outu, unsigned* __restrict__ mxu,
    float* __restrict__ den, const float* __restrict__ bias,
    float* __restrict__ h)
{
    int i = blockIdx.x * 256 + threadIdx.x;
    if (i >= N_NODES * 64) return;
    float v = ordVal(outu[i]) + bias[i & 63];
    h[i] = fmaxf(v, 0.0f);
    outu[i] = ORD_NEG;
    if (i < N_NODES) { mxu[i] = ORD_NEG; den[i] = 0.0f; }
}

// Final: out[n] = dot(h3[n], W4) + b4  (wave per node)
__global__ __launch_bounds__(256) void final_dot(
    const float* __restrict__ h3, const float* __restrict__ W4,
    const float* __restrict__ b4, float* __restrict__ out)
{
    int g = blockIdx.x * 256 + threadIdx.x;
    int n = g >> 5, lane = g & 31;
    if (n >= N_NODES) return;
    float2 hv = *(const float2*)(h3 + (size_t)n * 64 + lane * 2);
    float2 wv = *(const float2*)(W4 + lane * 2);
    float t = hv.x * wv.x + hv.y * wv.y;
    for (int o = 16; o > 0; o >>= 1) t += __shfl_xor(t, o);
    if (lane == 0) out[n] = t + b4[0];
}

// =====================================================================
extern "C" void kernel_launch(void* const* d_in, const int* in_sizes, int n_in,
                              void* d_out, int out_size, void* d_ws, size_t ws_size,
                              hipStream_t stream) {
    const float* x       = (const float*)d_in[0];
    const int*   ei      = (const int*)  d_in[1];
    const float* ea      = (const float*)d_in[2];
    const float* l1_Wl   = (const float*)d_in[3];
    const float* l1_bl   = (const float*)d_in[4];
    const float* l1_Wr   = (const float*)d_in[5];
    const float* l1_br   = (const float*)d_in[6];
    const float* l1_We   = (const float*)d_in[7];
    const float* l1_att  = (const float*)d_in[8];
    const float* l1_bias = (const float*)d_in[9];
    const float* l2_Wl   = (const float*)d_in[10];
    const float* l2_bl   = (const float*)d_in[11];
    const float* l2_Wr   = (const float*)d_in[12];
    const float* l2_br   = (const float*)d_in[13];
    const float* l2_We   = (const float*)d_in[14];
    const float* l2_att  = (const float*)d_in[15];
    const float* l2_bias = (const float*)d_in[16];
    const float* W3      = (const float*)d_in[17];
    const float* b3      = (const float*)d_in[18];
    const float* W4      = (const float*)d_in[19];
    const float* b4      = (const float*)d_in[20];

    const int* src = ei;
    const int* dst = ei + N_EDGES;

    // workspace carve-up (~110 MB + 100 KB packed weights)
    char* w = (char*)d_ws;
    float*    xl   = (float*)w;    w += (size_t)N_NODES * 64 * 4;
    float*    xr   = (float*)w;    w += (size_t)N_NODES * 64 * 4;
    float*    h    = (float*)w;    w += (size_t)N_NODES * 64 * 4;
    unsigned* outu = (unsigned*)w; w += (size_t)N_NODES * 64 * 4;
    float*    lg   = (float*)w;    w += (size_t)E_TOT * 4;          // logits -> ex
    unsigned* mxu  = (unsigned*)w; w += (size_t)N_NODES * 4;
    float*    den  = (float*)w;    w += (size_t)N_NODES * 4;
    float*    macc = (float*)w;    w += 256;
    __bf16* wb_l1l = (__bf16*)w;   w += (size_t)64 * KP1 * 2;       // packed bf16 weights
    __bf16* wb_l1r = (__bf16*)w;   w += (size_t)64 * KP1 * 2;
    __bf16* wb_l2l = (__bf16*)w;   w += (size_t)64 * 64 * 2;
    __bf16* wb_l2r = (__bf16*)w;   w += (size_t)64 * 64 * 2;
    __bf16* wb_w3  = (__bf16*)w;   w += (size_t)64 * 64 * 2;

    const dim3 B(256);
    const int GB_NODE   = (N_NODES * 64) / 256;    // 25000
    const int GB_EDGE_W = E_TOT / 8;               // 212500 (wave per edge)
    const int GB_EDGE_T = (E_TOT + 255) / 256;     // 6641
    const int GB_GEMM   = N_NODES / 16;            // 6250

    init_ws<<<GB_NODE, B, 0, stream>>>(outu, mxu, den, macc);
    mean_reduce<<<1024, B, 0, stream>>>(ea, macc);

    // pack all weights to bf16 (col-major, padded) once per launch
    pack_weight<<<(64 * KP1) / 256, B, 0, stream>>>(l1_Wl, D_IN, KP1, wb_l1l);
    pack_weight<<<(64 * KP1) / 256, B, 0, stream>>>(l1_Wr, D_IN, KP1, wb_l1r);
    pack_weight<<<(64 * 64) / 256, B, 0, stream>>>(l2_Wl, 64, 64, wb_l2l);
    pack_weight<<<(64 * 64) / 256, B, 0, stream>>>(l2_Wr, 64, 64, wb_l2r);
    pack_weight<<<(64 * 64) / 256, B, 0, stream>>>(W3, 64, 64, wb_w3);

    // ---- layer 1 ----
    gemm_dual<<<GB_GEMM, B, 0, stream>>>(x, D_IN, KP1,
                                         wb_l1l, l1_bl, xl, wb_l1r, l1_br, xr, 2, 0);
    edge_logits_max<<<GB_EDGE_W, B, 0, stream>>>(src, dst, ea, macc, xl, xr,
                                                 l1_We, l1_att, lg, mxu);
    edge_exp_sum<<<GB_EDGE_T, B, 0, stream>>>(dst, lg, mxu, den);
    edge_aggr_max<<<GB_EDGE_W, B, 0, stream>>>(src, dst, lg, den, xl, outu);
    node_finalize<<<GB_NODE, B, 0, stream>>>(outu, mxu, den, l1_bias, h);

    // ---- layer 2 ----
    gemm_dual<<<GB_GEMM, B, 0, stream>>>(h, 64, 64,
                                         wb_l2l, l2_bl, xl, wb_l2r, l2_br, xr, 2, 0);
    edge_logits_max<<<GB_EDGE_W, B, 0, stream>>>(src, dst, ea, macc, xl, xr,
                                                 l2_We, l2_att, lg, mxu);
    edge_exp_sum<<<GB_EDGE_T, B, 0, stream>>>(dst, lg, mxu, den);
    edge_aggr_max<<<GB_EDGE_W, B, 0, stream>>>(src, dst, lg, den, xl, outu);
    node_finalize<<<GB_NODE, B, 0, stream>>>(outu, mxu, den, l2_bias, h);

    // ---- MLP head: h3 = relu(h@W3 + b3); out = h3@W4 + b4 ----
    gemm_dual<<<GB_GEMM, B, 0, stream>>>(h, 64, 64,
                                         wb_w3, b3, xl, wb_w3, b3, xl, 1, 1);
    final_dot<<<(N_NODES * 32) / 256, B, 0, stream>>>(xl, W4, b4, (float*)d_out);
}
